// Geodesic_GNN_40845138985474
// MI455X (gfx1250) — matrix operand order repaired and checked by reading, not compiled, not run on hardware
//
#include <hip/hip_runtime.h>
#include <hip/hip_bf16.h>

// ============================================================================
// Geodesic GNN layer for MI455X (gfx1250, wave32, WMMA).
// All GEMMs use v_wmma_f32_16x16x32_bf16 (bf16 in, f32 accumulate).
// Input order assumption: setup_inputs() dict order, params flattened in
// insertion order (gv dicts: vec1, vec2, sca_w, gate_w, gate_b; tuples w,b).
// ============================================================================

typedef __bf16 bf16;
typedef __attribute__((ext_vector_type(16))) __bf16        v16bf;
typedef __attribute__((ext_vector_type(8)))  float         v8f;
typedef __attribute__((ext_vector_type(8)))  unsigned int  v8u;

#define DEVINL __device__ __forceinline__

static constexpr int NN = 20000;    // nodes (divisible by 16)
static constexpr int NE = 160000;   // edges (divisible by 16)

// ---------------------------------------------------------------------------
// WMMA fragment helpers (layouts per CDNA5 ISA 7.12.2)
// ---------------------------------------------------------------------------
DEVINL v8f vzero() {
  v8f c;
#pragma unroll
  for (int r = 0; r < 8; ++r) c[r] = 0.f;
  return c;
}

DEVINL v8f wmma_bf16(v16bf a, v16bf b, v8f c) {
  return __builtin_amdgcn_wmma_f32_16x16x32_bf16(false, a, false, b, (short)0, c,
                                                 false, false);
}

// A fragment (16x32 bf16) from LDS tile, row-major, row stride 'stride' elems.
// lane L holds A[m=L&15][k], k = kb*32 + (r>>2)*16 + (L>>4)*8 + (r&3)*2 + p
// (dwords r=0..3 and r=4..7 are contiguous -> 2x ds_load_b128)
DEVINL v16bf load_a_frag(const bf16* tile, int stride, int kb, int lane) {
  const int m = lane & 15, half = lane >> 4;
  const unsigned int* t32 = (const unsigned int*)tile;
  const int rowoff = (m * stride) >> 1;
  const int kbase  = (kb << 5) + (half << 3);
  v8u t;
#pragma unroll
  for (int r = 0; r < 8; ++r) {
    int k = kbase + ((r >> 2) << 4) + ((r & 3) << 1);
    t[r] = t32[rowoff + (k >> 1)];
  }
  return __builtin_bit_cast(v16bf, t);
}

// B fragment from pre-swizzled weights: 8 dwords contiguous per lane per frag.
DEVINL v16bf load_b_frag(const unsigned int* w, int f, int lane) {
  v8u t = *(const v8u*)(w + (((unsigned)f << 5) + (unsigned)lane) * 8u);
  return __builtin_bit_cast(v16bf, t);
}

// D fragment element (m = r + 8*(L>>4), n = L&15)
DEVINL void store_d_bf16(bf16* tile, int stride, int ncol0, v8f d, int lane) {
  const int n = lane & 15, half = lane >> 4;
#pragma unroll
  for (int r = 0; r < 8; ++r) tile[(r + half * 8) * stride + ncol0 + n] = (bf16)d[r];
}
DEVINL void store_d_f32(float* tile, int stride, int ncol0, v8f d, int lane) {
  const int n = lane & 15, half = lane >> 4;
#pragma unroll
  for (int r = 0; r < 8; ++r) tile[(r + half * 8) * stride + ncol0 + n] = d[r];
}

DEVINL v8f bias_frag(const float* b, int nt, int lane) {
  float v = b ? b[nt * 16 + (lane & 15)] : 0.f;
  v8f c;
#pragma unroll
  for (int r = 0; r < 8; ++r) c[r] = v;
  return c;
}

DEVINL float sigmoidf(float x) { return 1.f / (1.f + __expf(-x)); }

// ---------------------------------------------------------------------------
// Generic GVLinear (in_s=256, in_v=64 -> out_s=256, out_v=64, h=64)
// ---------------------------------------------------------------------------
struct GvW { const unsigned int *v1, *sca, *gate, *v2; const float* gate_b; };

// S:   16x320 bf16 LDS (cols 0..255 = input scalars, fn fills 256..319 = |vi|)
// V:   3 x 16x64 bf16 LDS input vectors
// VI:  3 x 16x64 bf16 LDS scratch (vi)
// OUTS:16x256 bf16 LDS (scalar output, A-staging for gate)
DEVINL void gv256_64(bf16* S, const bf16* V, bf16* VI, bf16* OUTS, GvW w,
                     float* sca_g, float* vec_g, bf16* vec_lds, int row0, int lane) {
  const int n = lane & 15, half = lane >> 4;
  // ---- phase 1: vi = vec1 @ V (per d), vn = |vi| into S cols 256..319 ----
  {
    v16bf av[3][2];
#pragma unroll
    for (int d = 0; d < 3; ++d) {
      av[d][0] = load_a_frag(V + d * 1024, 64, 0, lane);
      av[d][1] = load_a_frag(V + d * 1024, 64, 1, lane);
    }
    for (int nt = 0; nt < 4; ++nt) {
      v8f ss = vzero();
#pragma unroll
      for (int d = 0; d < 3; ++d) {
        v8f c = vzero();
        c = wmma_bf16(av[d][0], load_b_frag(w.v1, nt * 2 + 0, lane), c);
        c = wmma_bf16(av[d][1], load_b_frag(w.v1, nt * 2 + 1, lane), c);
        store_d_bf16(VI + d * 1024, 64, nt * 16, c, lane);
        ss += c * c;
      }
      v8f vn;
#pragma unroll
      for (int r = 0; r < 8; ++r) vn[r] = __builtin_sqrtf(ss[r]);
      store_d_bf16(S, 320, 256 + nt * 16, vn, lane);
    }
  }
  // ---- phase 2: sca_out = concat(S) @ sca_w^T  (K=320, N=256) ----
  {
    v16bf as[10];
#pragma unroll
    for (int kt = 0; kt < 10; ++kt) as[kt] = load_a_frag(S, 320, kt, lane);
    for (int nt = 0; nt < 16; ++nt) {
      v8f c = vzero();
#pragma unroll
      for (int kt = 0; kt < 10; ++kt)
        c = wmma_bf16(as[kt], load_b_frag(w.sca, nt * 10 + kt, lane), c);
      store_d_bf16(OUTS, 256, nt * 16, c, lane);
      if (sca_g) {
#pragma unroll
        for (int r = 0; r < 8; ++r)
          sca_g[(size_t)(row0 + r + half * 8) * 256 + nt * 16 + n] = c[r];
      }
    }
  }
  // ---- phase 3: gate = sigmoid(sca_out @ gate_w^T + b); vec = gate*(vec2@vi) ----
  {
    v16bf ag[8];
#pragma unroll
    for (int kt = 0; kt < 8; ++kt) ag[kt] = load_a_frag(OUTS, 256, kt, lane);
    v16bf avi[3][2];
#pragma unroll
    for (int d = 0; d < 3; ++d) {
      avi[d][0] = load_a_frag(VI + d * 1024, 64, 0, lane);
      avi[d][1] = load_a_frag(VI + d * 1024, 64, 1, lane);
    }
    for (int nt = 0; nt < 4; ++nt) {
      v8f g = bias_frag(w.gate_b, nt, lane);
#pragma unroll
      for (int kt = 0; kt < 8; ++kt)
        g = wmma_bf16(ag[kt], load_b_frag(w.gate, nt * 8 + kt, lane), g);
#pragma unroll
      for (int r = 0; r < 8; ++r) g[r] = sigmoidf(g[r]);
#pragma unroll
      for (int d = 0; d < 3; ++d) {
        v8f c = vzero();
        c = wmma_bf16(avi[d][0], load_b_frag(w.v2, nt * 2 + 0, lane), c);
        c = wmma_bf16(avi[d][1], load_b_frag(w.v2, nt * 2 + 1, lane), c);
        c *= g;
        if (vec_lds) store_d_bf16(vec_lds + d * 1024, 64, nt * 16, c, lane);
        if (vec_g) {
#pragma unroll
          for (int r = 0; r < 8; ++r)
            vec_g[(size_t)(row0 + r + half * 8) * 192 + (nt * 16 + n) * 3 + d] = c[r];
        }
      }
    }
  }
}

// ---------------------------------------------------------------------------
// Prep kernels: fold edge-mapper constants; swizzle weights to B-frag order
// ---------------------------------------------------------------------------
__global__ void gnn_prep_const(const float* v1e, const float* v2e, const float* wexp,
                               const float* scawe, float* c1g, float* c2g, float* cbg) {
  __shared__ float sc1[64], sa1[64];
  int t = threadIdx.x;  // 256 threads
  if (t < 64) {
    float s = 0.f;
    for (int v = 0; v < 64; ++v) s += v1e[t * 64 + v] * wexp[v];
    sc1[t] = s; sa1[t] = fabsf(s); c1g[t] = s;
  }
  __syncthreads();
  if (t < 64) {
    float s = 0.f;
    for (int h = 0; h < 64; ++h) s += v2e[t * 64 + h] * sc1[h];
    c2g[t] = s;
  }
  float s = 0.f;
  for (int h = 0; h < 64; ++h) s += scawe[t * 128 + 64 + h] * sa1[h];
  cbg[t] = s;
}

// Wt[k][n] = src[n*srs + k]; dst dword idx = ((nt*KT+kt)*32 + lane)*8 + r,
// packing bf16 pair (k0,k0+1) with k0 = kt*32 + (r>>2)*16 + (lane>>4)*8 + (r&3)*2.
__global__ void gnn_swizzle(const float* src, unsigned int* dst, int K, int N, int srs) {
  int idx = blockIdx.x * blockDim.x + threadIdx.x;
  int total = (K * N) >> 1;
  if (idx >= total) return;
  int r = idx & 7, lane = (idx >> 3) & 31, frag = idx >> 8;
  int KT = K >> 5;
  int nt = frag / KT, kt = frag - nt * KT;
  int n  = nt * 16 + (lane & 15);
  int k0 = kt * 32 + ((r >> 2) << 4) + ((lane >> 4) << 3) + ((r & 3) << 1);
  bf16 lo = (bf16)src[(size_t)n * srs + k0];
  bf16 hi = (bf16)src[(size_t)n * srs + k0 + 1];
  unsigned int u = (unsigned int)__builtin_bit_cast(unsigned short, lo) |
                   ((unsigned int)__builtin_bit_cast(unsigned short, hi) << 16);
  dst[idx] = u;
}

// ---------------------------------------------------------------------------
// Kernel 1: per-node precompute (node_mapper -> a_ss/a_sv/a_vv; resi -> aggr)
// ---------------------------------------------------------------------------
struct NodeArgs {
  const float *node_sca, *node_vec;
  const unsigned int *W_NV1, *W_NS, *W_NG, *W_NV2;
  const unsigned int *W_ASS, *W_ASV, *W_AVV;
  const unsigned int *W_RV1, *W_RS, *W_RG, *W_RV2;
  const float *b_NG, *b_ASS, *b_ASV, *b_RG;
  bf16 *a_ss, *a_sv, *a_vv;
  float *aggr_sca, *aggr_vec;
};

struct NodeSmem {
  bf16 S[16 * 320];
  bf16 V[3 * 16 * 64];
  bf16 VI[3 * 16 * 64];
  bf16 OUTS[16 * 256];
  bf16 NV[3 * 16 * 64];
};

__global__ void __launch_bounds__(128) gnn_node(NodeArgs a) {
  __shared__ NodeSmem sm[4];
  const int wv = threadIdx.x >> 5, lane = threadIdx.x & 31;
  const int tile = blockIdx.x * 4 + wv;
  if (tile >= NN / 16) return;
  const int row0 = tile * 16;
  NodeSmem& s = sm[wv];
  const int n = lane & 15, half = lane >> 4;

  for (int idx = lane; idx < 16 * 256; idx += 32) {
    int r = idx >> 8, c = idx & 255;
    s.S[r * 320 + c] = (bf16)a.node_sca[(size_t)(row0 + r) * 256 + c];
  }
  for (int idx = lane; idx < 3 * 16 * 64; idx += 32) {
    int d = idx >> 10, rem = idx & 1023, r = rem >> 6, c = rem & 63;
    s.V[d * 1024 + r * 64 + c] = (bf16)a.node_vec[(size_t)(row0 + r) * 192 + c * 3 + d];
  }

  GvW nm{a.W_NV1, a.W_NS, a.W_NG, a.W_NV2, a.b_NG};
  gv256_64(s.S, s.V, s.VI, s.OUTS, nm, nullptr, nullptr, s.NV, row0, lane);

  // a_ss / a_sv : (16x256) @ (256x128) + b  -> bf16 gather tables
  {
    v16bf an[8];
#pragma unroll
    for (int kt = 0; kt < 8; ++kt) an[kt] = load_a_frag(s.OUTS, 256, kt, lane);
    for (int nt = 0; nt < 8; ++nt) {
      v8f c = bias_frag(a.b_ASS, nt, lane);
#pragma unroll
      for (int kt = 0; kt < 8; ++kt)
        c = wmma_bf16(an[kt], load_b_frag(a.W_ASS, nt * 8 + kt, lane), c);
#pragma unroll
      for (int r = 0; r < 8; ++r)
        a.a_ss[(size_t)(row0 + r + half * 8) * 128 + nt * 16 + n] = (bf16)c[r];
    }
    for (int nt = 0; nt < 8; ++nt) {
      v8f c = bias_frag(a.b_ASV, nt, lane);
#pragma unroll
      for (int kt = 0; kt < 8; ++kt)
        c = wmma_bf16(an[kt], load_b_frag(a.W_ASV, nt * 8 + kt, lane), c);
#pragma unroll
      for (int r = 0; r < 8; ++r)
        a.a_sv[(size_t)(row0 + r + half * 8) * 128 + nt * 16 + n] = (bf16)c[r];
    }
    // a_vv : per d, (16x64 nvec) @ (64x128)
    v16bf av[3][2];
#pragma unroll
    for (int d = 0; d < 3; ++d) {
      av[d][0] = load_a_frag(s.NV + d * 1024, 64, 0, lane);
      av[d][1] = load_a_frag(s.NV + d * 1024, 64, 1, lane);
    }
#pragma unroll
    for (int d = 0; d < 3; ++d)
      for (int nt = 0; nt < 8; ++nt) {
        v8f c = vzero();
        c = wmma_bf16(av[d][0], load_b_frag(a.W_AVV, nt * 2 + 0, lane), c);
        c = wmma_bf16(av[d][1], load_b_frag(a.W_AVV, nt * 2 + 1, lane), c);
#pragma unroll
        for (int r = 0; r < 8; ++r)
          a.a_vv[(size_t)(row0 + r + half * 8) * 384 + (nt * 16 + n) * 3 + d] = (bf16)c[r];
      }
  }
  // resi -> aggregation buffers (init value for segment-sum)
  GvW rw{a.W_RV1, a.W_RS, a.W_RG, a.W_RV2, a.b_RG};
  gv256_64(s.S, s.V, s.VI, s.OUTS, rw, a.aggr_sca, a.aggr_vec, nullptr, row0, lane);
}

// ---------------------------------------------------------------------------
// Kernel 2: fused per-edge pipeline (1 wave = 16 edges)
// ---------------------------------------------------------------------------
struct EdgeArgs {
  const float *edge_feat, *edge_vec, *gds;
  const int* edge_row;
  const unsigned int *W_ES, *W_GE, *W_SSE, *W_SVE, *W_VVE;
  const unsigned int *W_MV1, *W_MS, *W_MV2, *W_MG;
  const float *cb, *c2, *b_GE, *b_SSE, *b_SVE, *b_MG;
  const bf16 *a_ss, *a_sv, *a_vv;
  float *aggr_sca, *aggr_vec;
};

struct EdgeSmem {
  bf16  bufA[16 * 256];      // edge_sca | g2 | m_sca staging
  bf16  bufB[16 * 256];      // esca | concat(msg_sca, vnorm)
  float f0[16 * 128];        // e_sv (f32)
  float f1[16 * 128];        // e_vv (f32)
  bf16  mv[3 * 16 * 128];    // msg_vec bf16 (A for vec1)
  bf16  vi[3 * 16 * 128];    // vi bf16 (A for vec2)
  float dist[16];
  float coeff[16];
  float ev[16][3];
  int   nidx[16];
};

__global__ void __launch_bounds__(128) gnn_edge(EdgeArgs a) {
  __shared__ EdgeSmem sm[4];
  const int wv = threadIdx.x >> 5, lane = threadIdx.x & 31;
  const int tile = blockIdx.x * 4 + wv;
  if (tile >= NE / 16) return;
  const int e0 = tile * 16;
  EdgeSmem& s = sm[wv];
  const int n = lane & 15, half = lane >> 4;

  // ---- per-edge meta ----
  if (lane < 16) {
    int e = e0 + lane;
    float d = a.gds[e];
    s.dist[lane] = d;
    float co = 0.5f * (__cosf(d * 0.31415926535897931f) + 1.f);
    s.coeff[lane] = (d >= 0.f && d <= 10.f) ? co : 0.f;
    float vx = a.edge_vec[e * 3 + 0], vy = a.edge_vec[e * 3 + 1], vz = a.edge_vec[e * 3 + 2];
    float inv = 1.f / (__builtin_sqrtf(vx * vx + vy * vy + vz * vz) + 1e-7f);
    s.ev[lane][0] = vx * inv; s.ev[lane][1] = vy * inv; s.ev[lane][2] = vz * inv;
    s.nidx[lane] = a.edge_row[e];
  }
  // ---- prefetch gathered node rows (L2 latency hidden behind GEMM1..5) ----
  // 10 cachelines (128B) per edge: a_ss row 256B, a_sv row 256B, a_vv row 768B.
  // One global_prefetch_b8 covers 32 lane addresses -> 5 instructions total.
  for (int t = lane; t < 160; t += 32) {
    int e = t / 10, part = t - (t / 10) * 10;
    int ni = s.nidx[e];
    const char* p;
    if (part < 2)      p = (const char*)(a.a_ss + (size_t)ni * 128) + part * 128;
    else if (part < 4) p = (const char*)(a.a_sv + (size_t)ni * 128) + (part - 2) * 128;
    else               p = (const char*)(a.a_vv + (size_t)ni * 384) + (part - 4) * 128;
    __builtin_prefetch(p, 0, 0);
  }
  // ---- edge_sca = [gauss(62), feat(2)] into bufA (stride 64) ----
  const float step = 10.f / 61.f;
  const float gco  = -0.5f / (step * step);
  for (int idx = lane; idx < 16 * 64; idx += 32) {
    int r = idx >> 6, c = idx & 63;
    float v;
    if (c < 62) { float dd = s.dist[r] - step * (float)c; v = __expf(gco * dd * dd); }
    else        { v = a.edge_feat[(size_t)(e0 + r) * 2 + (c - 62)]; }
    s.bufA[r * 64 + c] = (bf16)v;
  }
  // ---- GEMM1: esca = edge_sca @ W_ES + cb  (16x64 @ 64x256) -> bufB ----
  {
    v16bf a0 = load_a_frag(s.bufA, 64, 0, lane), a1 = load_a_frag(s.bufA, 64, 1, lane);
    for (int nt = 0; nt < 16; ++nt) {
      v8f c = bias_frag(a.cb, nt, lane);
      c = wmma_bf16(a0, load_b_frag(a.W_ES, nt * 2 + 0, lane), c);
      c = wmma_bf16(a1, load_b_frag(a.W_ES, nt * 2 + 1, lane), c);
      store_d_bf16(s.bufB, 256, nt * 16, c, lane);
    }
  }
  v16bf aes[8];
#pragma unroll
  for (int kt = 0; kt < 8; ++kt) aes[kt] = load_a_frag(s.bufB, 256, kt, lane);

  // ---- GEMM2: g2 = sigmoid(esca@gate_w^T+b) * c2 -> bufA (stride 64) ----
  for (int nt = 0; nt < 4; ++nt) {
    v8f g = bias_frag(a.b_GE, nt, lane);
#pragma unroll
    for (int kt = 0; kt < 8; ++kt)
      g = wmma_bf16(aes[kt], load_b_frag(a.W_GE, nt * 8 + kt, lane), g);
    float c2v = a.c2[nt * 16 + n];
#pragma unroll
    for (int r = 0; r < 8; ++r) g[r] = c2v * sigmoidf(g[r]);
    store_d_bf16(s.bufA, 64, nt * 16, g, lane);
  }
  // ---- GEMM3: e_ss; fuse msg_sca = e_ss * a_ss[row] * coeff -> bufB[0:128] ----
  for (int nt = 0; nt < 8; ++nt) {
    v8f c = bias_frag(a.b_SSE, nt, lane);
#pragma unroll
    for (int kt = 0; kt < 8; ++kt)
      c = wmma_bf16(aes[kt], load_b_frag(a.W_SSE, nt * 8 + kt, lane), c);
#pragma unroll
    for (int r = 0; r < 8; ++r) {
      int m = r + half * 8;
      int ni = s.nidx[m];
      float av = (float)a.a_ss[(size_t)ni * 128 + nt * 16 + n];
      s.bufB[m * 256 + nt * 16 + n] = (bf16)(c[r] * av * s.coeff[m]);
    }
  }
  // ---- GEMM4: e_sv (f32 -> f0) ----
  for (int nt = 0; nt < 8; ++nt) {
    v8f c = bias_frag(a.b_SVE, nt, lane);
#pragma unroll
    for (int kt = 0; kt < 8; ++kt)
      c = wmma_bf16(aes[kt], load_b_frag(a.W_SVE, nt * 8 + kt, lane), c);
    store_d_f32(s.f0, 128, nt * 16, c, lane);
  }
  // ---- GEMM5: e_vv = g2 @ W_VVE (16x64 @ 64x128) -> f1 ----
  {
    v16bf g0 = load_a_frag(s.bufA, 64, 0, lane), g1 = load_a_frag(s.bufA, 64, 1, lane);
    for (int nt = 0; nt < 8; ++nt) {
      v8f c = vzero();
      c = wmma_bf16(g0, load_b_frag(a.W_VVE, nt * 2 + 0, lane), c);
      c = wmma_bf16(g1, load_b_frag(a.W_VVE, nt * 2 + 1, lane), c);
      store_d_f32(s.f1, 128, nt * 16, c, lane);
    }
  }
  // ---- msg_vec build (gathers a_vv / a_sv from L2-resident tables) ----
  for (int d = 0; d < 3; ++d)
    for (int idx = lane; idx < 16 * 128; idx += 32) {
      int r = idx >> 7, h = idx & 127;
      int ni = s.nidx[r];
      float esv = s.f0[r * 128 + h], evv = s.f1[r * 128 + h];
      float avv = (float)a.a_vv[(size_t)ni * 384 + h * 3 + d];
      float asv = (float)a.a_sv[(size_t)ni * 128 + h];
      float mvv = (avv * esv + asv * evv * s.ev[r][d]) * s.coeff[r];
      s.mv[d * 2048 + r * 128 + h] = (bf16)mvv;
    }
  // ---- msg_out GV: vec1 (3x 16x128@128x128) + vnorm -> bufB[128:256] ----
  {
    v16bf amv[3][4];
#pragma unroll
    for (int d = 0; d < 3; ++d)
#pragma unroll
      for (int kt = 0; kt < 4; ++kt)
        amv[d][kt] = load_a_frag(s.mv + d * 2048, 128, kt, lane);
    for (int nt = 0; nt < 8; ++nt) {
      v8f ss = vzero();
#pragma unroll
      for (int d = 0; d < 3; ++d) {
        v8f c = vzero();
#pragma unroll
        for (int kt = 0; kt < 4; ++kt)
          c = wmma_bf16(amv[d][kt], load_b_frag(a.W_MV1, nt * 4 + kt, lane), c);
        store_d_bf16(s.vi + d * 2048, 128, nt * 16, c, lane);
        ss += c * c;
      }
      v8f vn;
#pragma unroll
      for (int r = 0; r < 8; ++r) vn[r] = __builtin_sqrtf(ss[r]);
      store_d_bf16(s.bufB, 256, 128 + nt * 16, vn, lane);
    }
  }
  // ---- m_sca = concat @ sca_w^T (16x256@256x256): atomic into aggr_sca ----
  {
    v16bf ac[8];
#pragma unroll
    for (int kt = 0; kt < 8; ++kt) ac[kt] = load_a_frag(s.bufB, 256, kt, lane);
    for (int nt = 0; nt < 16; ++nt) {
      v8f c = vzero();
#pragma unroll
      for (int kt = 0; kt < 8; ++kt)
        c = wmma_bf16(ac[kt], load_b_frag(a.W_MS, nt * 8 + kt, lane), c);
      store_d_bf16(s.bufA, 256, nt * 16, c, lane);
#pragma unroll
      for (int r = 0; r < 8; ++r) {
        int m = r + half * 8;
        atomicAdd(&a.aggr_sca[(size_t)s.nidx[m] * 256 + nt * 16 + n], c[r]);
      }
    }
  }
  // ---- gate + vec2: msg_vec_out = sigmoid(gate) * (vec2@vi): atomic aggr_vec ----
  {
    v16bf ao[8];
#pragma unroll
    for (int kt = 0; kt < 8; ++kt) ao[kt] = load_a_frag(s.bufA, 256, kt, lane);
    for (int nt = 0; nt < 4; ++nt) {
      v8f g = bias_frag(a.b_MG, nt, lane);
#pragma unroll
      for (int kt = 0; kt < 8; ++kt)
        g = wmma_bf16(ao[kt], load_b_frag(a.W_MG, nt * 8 + kt, lane), g);
#pragma unroll
      for (int r = 0; r < 8; ++r) g[r] = sigmoidf(g[r]);
#pragma unroll
      for (int d = 0; d < 3; ++d) {
        v16bf t0 = load_a_frag(s.vi + d * 2048, 128, 0, lane);
        v16bf t1 = load_a_frag(s.vi + d * 2048, 128, 1, lane);
        v16bf t2 = load_a_frag(s.vi + d * 2048, 128, 2, lane);
        v16bf t3 = load_a_frag(s.vi + d * 2048, 128, 3, lane);
        v8f c = vzero();
        c = wmma_bf16(t0, load_b_frag(a.W_MV2, nt * 4 + 0, lane), c);
        c = wmma_bf16(t1, load_b_frag(a.W_MV2, nt * 4 + 1, lane), c);
        c = wmma_bf16(t2, load_b_frag(a.W_MV2, nt * 4 + 2, lane), c);
        c = wmma_bf16(t3, load_b_frag(a.W_MV2, nt * 4 + 3, lane), c);
#pragma unroll
        for (int r = 0; r < 8; ++r) {
          int m = r + half * 8;
          atomicAdd(&a.aggr_vec[(size_t)s.nidx[m] * 192 + (nt * 16 + n) * 3 + d],
                    c[r] * g[r]);
        }
      }
    }
  }
}

// ---------------------------------------------------------------------------
// Kernel 3: aggr_out GVLinear on (resi + aggregated messages) -> d_out
// ---------------------------------------------------------------------------
struct OutArgs {
  const float *aggr_sca, *aggr_vec;
  const unsigned int *W_OV1, *W_OS, *W_OG, *W_OV2;
  const float* b_OG;
  float *out_sca, *out_vec;
};

struct OutSmem {
  bf16 S[16 * 320];
  bf16 V[3 * 16 * 64];
  bf16 VI[3 * 16 * 64];
  bf16 OUTS[16 * 256];
};

__global__ void __launch_bounds__(128) gnn_out(OutArgs a) {
  __shared__ OutSmem sm[4];
  const int wv = threadIdx.x >> 5, lane = threadIdx.x & 31;
  const int tile = blockIdx.x * 4 + wv;
  if (tile >= NN / 16) return;
  const int row0 = tile * 16;
  OutSmem& s = sm[wv];
  for (int idx = lane; idx < 16 * 256; idx += 32) {
    int r = idx >> 8, c = idx & 255;
    s.S[r * 320 + c] = (bf16)a.aggr_sca[(size_t)(row0 + r) * 256 + c];
  }
  for (int idx = lane; idx < 3 * 16 * 64; idx += 32) {
    int d = idx >> 10, rem = idx & 1023, r = rem >> 6, c = rem & 63;
    s.V[d * 1024 + r * 64 + c] = (bf16)a.aggr_vec[(size_t)(row0 + r) * 192 + c * 3 + d];
  }
  GvW ow{a.W_OV1, a.W_OS, a.W_OG, a.W_OV2, a.b_OG};
  gv256_64(s.S, s.V, s.VI, s.OUTS, ow, a.out_sca, a.out_vec, nullptr, row0, lane);
}

// ---------------------------------------------------------------------------
// Host launcher
// ---------------------------------------------------------------------------
static void launch_swz(const float* src, unsigned int* dst, int K, int N, int srs,
                       hipStream_t stream) {
  int total = (K * N) >> 1;
  gnn_swizzle<<<(total + 255) / 256, 256, 0, stream>>>(src, dst, K, N, srs);
}

extern "C" void kernel_launch(void* const* d_in, const int* in_sizes, int n_in,
                              void* d_out, int out_size, void* d_ws, size_t ws_size,
                              hipStream_t stream) {
  (void)in_sizes; (void)n_in; (void)out_size; (void)ws_size;
  const float* in_node_sca  = (const float*)d_in[0];
  const float* in_node_vec  = (const float*)d_in[1];
  const float* in_edge_feat = (const float*)d_in[2];
  const float* in_edge_vec  = (const float*)d_in[3];
  const float* in_gds       = (const float*)d_in[4];
  const int*   in_edge_row  = (const int*)d_in[5];  // edge_index[0] = first NE ints
  const float* p_edge_exp_w = (const float*)d_in[6];
  const float* p_nm_vec1  = (const float*)d_in[7];
  const float* p_nm_vec2  = (const float*)d_in[8];
  const float* p_nm_scaw  = (const float*)d_in[9];
  const float* p_nm_gatew = (const float*)d_in[10];
  const float* p_nm_gateb = (const float*)d_in[11];
  const float* p_em_vec1  = (const float*)d_in[12];
  const float* p_em_vec2  = (const float*)d_in[13];
  const float* p_em_scaw  = (const float*)d_in[14];
  const float* p_em_gatew = (const float*)d_in[15];
  const float* p_em_gateb = (const float*)d_in[16];
  const float* p_ess_w = (const float*)d_in[17];
  const float* p_ess_b = (const float*)d_in[18];
  const float* p_nss_w = (const float*)d_in[19];
  const float* p_nss_b = (const float*)d_in[20];
  const float* p_esv_w = (const float*)d_in[21];
  const float* p_esv_b = (const float*)d_in[22];
  const float* p_nsv_w = (const float*)d_in[23];
  const float* p_nsv_b = (const float*)d_in[24];
  const float* p_evv   = (const float*)d_in[25];
  const float* p_nvv   = (const float*)d_in[26];
  const float* p_mo_vec1  = (const float*)d_in[27];
  const float* p_mo_vec2  = (const float*)d_in[28];
  const float* p_mo_scaw  = (const float*)d_in[29];
  const float* p_mo_gatew = (const float*)d_in[30];
  const float* p_mo_gateb = (const float*)d_in[31];
  const float* p_r_vec1  = (const float*)d_in[32];
  const float* p_r_vec2  = (const float*)d_in[33];
  const float* p_r_scaw  = (const float*)d_in[34];
  const float* p_r_gatew = (const float*)d_in[35];
  const float* p_r_gateb = (const float*)d_in[36];
  const float* p_o_vec1  = (const float*)d_in[37];
  const float* p_o_vec2  = (const float*)d_in[38];
  const float* p_o_scaw  = (const float*)d_in[39];
  const float* p_o_gatew = (const float*)d_in[40];
  const float* p_o_gateb = (const float*)d_in[41];

  // ---- workspace carve ----
  char* wsb = (char*)d_ws;
  size_t off = 0;
  auto carve = [&](size_t bytes) -> void* {
    void* p = wsb + off;
    off = (off + bytes + 255) & ~(size_t)255;
    return p;
  };
  auto wz = [&](int K, int N) { return (unsigned int*)carve((size_t)K * N * 2); };

  unsigned int* W_ES  = wz(64, 256);
  unsigned int* W_GE  = wz(256, 64);
  unsigned int* W_SSE = wz(256, 128);
  unsigned int* W_SVE = wz(256, 128);
  unsigned int* W_VVE = wz(64, 128);
  unsigned int* W_MV1 = wz(128, 128);
  unsigned int* W_MS  = wz(256, 256);
  unsigned int* W_MV2 = wz(128, 64);
  unsigned int* W_MG  = wz(256, 64);
  unsigned int* W_NV1 = wz(64, 64);
  unsigned int* W_NS  = wz(320, 256);
  unsigned int* W_NG  = wz(256, 64);
  unsigned int* W_NV2 = wz(64, 64);
  unsigned int* W_ASS = wz(256, 128);
  unsigned int* W_ASV = wz(256, 128);
  unsigned int* W_AVV = wz(64, 128);
  unsigned int* W_RV1 = wz(64, 64);
  unsigned int* W_RS  = wz(320, 256);
  unsigned int* W_RG  = wz(256, 64);
  unsigned int* W_RV2 = wz(64, 64);
  unsigned int* W_OV1 = wz(64, 64);
  unsigned int* W_OS  = wz(320, 256);
  unsigned int* W_OG  = wz(256, 64);
  unsigned int* W_OV2 = wz(64, 64);
  float* c1  = (float*)carve(64 * 4);
  float* c2  = (float*)carve(64 * 4);
  float* cb  = (float*)carve(256 * 4);
  bf16*  a_ss = (bf16*)carve((size_t)NN * 128 * 2);
  bf16*  a_sv = (bf16*)carve((size_t)NN * 128 * 2);
  bf16*  a_vv = (bf16*)carve((size_t)NN * 384 * 2);
  float* aggr_sca = (float*)carve((size_t)NN * 256 * 4);
  float* aggr_vec = (float*)carve((size_t)NN * 192 * 4);

  // ---- prep: constants + weight swizzles (B-fragment order, bf16) ----
  gnn_prep_const<<<1, 256, 0, stream>>>(p_em_vec1, p_em_vec2, p_edge_exp_w,
                                        p_em_scaw, c1, c2, cb);
  launch_swz(p_em_scaw, W_ES, 64, 256, 128, stream);   // first 64 in-cols of edge sca_w
  launch_swz(p_em_gatew, W_GE, 256, 64, 256, stream);
  launch_swz(p_ess_w, W_SSE, 256, 128, 256, stream);
  launch_swz(p_esv_w, W_SVE, 256, 128, 256, stream);
  launch_swz(p_evv, W_VVE, 64, 128, 64, stream);
  launch_swz(p_mo_vec1, W_MV1, 128, 128, 128, stream);
  launch_swz(p_mo_scaw, W_MS, 256, 256, 256, stream);
  launch_swz(p_mo_vec2, W_MV2, 128, 64, 128, stream);
  launch_swz(p_mo_gatew, W_MG, 256, 64, 256, stream);
  launch_swz(p_nm_vec1, W_NV1, 64, 64, 64, stream);
  launch_swz(p_nm_scaw, W_NS, 320, 256, 320, stream);
  launch_swz(p_nm_gatew, W_NG, 256, 64, 256, stream);
  launch_swz(p_nm_vec2, W_NV2, 64, 64, 64, stream);
  launch_swz(p_nss_w, W_ASS, 256, 128, 256, stream);
  launch_swz(p_nsv_w, W_ASV, 256, 128, 256, stream);
  launch_swz(p_nvv, W_AVV, 64, 128, 64, stream);
  launch_swz(p_r_vec1, W_RV1, 64, 64, 64, stream);
  launch_swz(p_r_scaw, W_RS, 320, 256, 320, stream);
  launch_swz(p_r_gatew, W_RG, 256, 64, 256, stream);
  launch_swz(p_r_vec2, W_RV2, 64, 64, 64, stream);
  launch_swz(p_o_vec1, W_OV1, 64, 64, 64, stream);
  launch_swz(p_o_scaw, W_OS, 320, 256, 320, stream);
  launch_swz(p_o_gatew, W_OG, 256, 64, 256, stream);
  launch_swz(p_o_vec2, W_OV2, 64, 64, 64, stream);

  // ---- node precompute + resi init ----
  NodeArgs na;
  na.node_sca = in_node_sca; na.node_vec = in_node_vec;
  na.W_NV1 = W_NV1; na.W_NS = W_NS; na.W_NG = W_NG; na.W_NV2 = W_NV2;
  na.W_ASS = W_ASS; na.W_ASV = W_ASV; na.W_AVV = W_AVV;
  na.W_RV1 = W_RV1; na.W_RS = W_RS; na.W_RG = W_RG; na.W_RV2 = W_RV2;
  na.b_NG = p_nm_gateb; na.b_ASS = p_nss_b; na.b_ASV = p_nsv_b; na.b_RG = p_r_gateb;
  na.a_ss = a_ss; na.a_sv = a_sv; na.a_vv = a_vv;
  na.aggr_sca = aggr_sca; na.aggr_vec = aggr_vec;
  gnn_node<<<(NN / 16 + 3) / 4, 128, 0, stream>>>(na);

  // ---- edge pipeline ----
  EdgeArgs ea;
  ea.edge_feat = in_edge_feat; ea.edge_vec = in_edge_vec; ea.gds = in_gds;
  ea.edge_row = in_edge_row;
  ea.W_ES = W_ES; ea.W_GE = W_GE; ea.W_SSE = W_SSE; ea.W_SVE = W_SVE; ea.W_VVE = W_VVE;
  ea.W_MV1 = W_MV1; ea.W_MS = W_MS; ea.W_MV2 = W_MV2; ea.W_MG = W_MG;
  ea.cb = cb; ea.c2 = c2; ea.b_GE = p_em_gateb; ea.b_SSE = p_ess_b;
  ea.b_SVE = p_esv_b; ea.b_MG = p_mo_gateb;
  ea.a_ss = a_ss; ea.a_sv = a_sv; ea.a_vv = a_vv;
  ea.aggr_sca = aggr_sca; ea.aggr_vec = aggr_vec;
  gnn_edge<<<NE / 16 / 4, 128, 0, stream>>>(ea);

  // ---- final aggr_out -> d_out ----
  OutArgs oa;
  oa.aggr_sca = aggr_sca; oa.aggr_vec = aggr_vec;
  oa.W_OV1 = W_OV1; oa.W_OS = W_OS; oa.W_OG = W_OG; oa.W_OV2 = W_OV2;
  oa.b_OG = p_o_gateb;
  oa.out_sca = (float*)d_out;
  oa.out_vec = (float*)d_out + (size_t)NN * 256;
  gnn_out<<<(NN / 16 + 3) / 4, 128, 0, stream>>>(oa);
}
